// GlobalSS2DBranch_78219944394838
// MI455X (gfx1250) — matrix operand order repaired and checked by reading, not compiled, not run on hardware
//
#include <hip/hip_runtime.h>
#include <hip/hip_bf16.h>
#include <cstdint>

typedef _Float16 half_t;
typedef __attribute__((ext_vector_type(16))) _Float16 v16h;
typedef __attribute__((ext_vector_type(8)))  _Float16 v8h;
typedef __attribute__((ext_vector_type(8)))  float    v8f;

#define B_    8
#define C_    128
#define Dd    128
#define Nn    16
#define Rr    8
#define Ls    4096
#define ROWS  32768              // B_ * Ls
#define NDIR  4
#define XPN   48                 // x_dbl padded width (R + 2N = 40 -> 48)

// ---------------------------------------------------------------------------
// WMMA A fragment (16x32 f16) from row-major [*, lda], per ISA 7.12.2:
// lane<16: rows 0..15, K {0..7,16..23}; lane>=16: K {8..15,24..31}.
// ---------------------------------------------------------------------------
static __device__ __forceinline__ v16h load_a_frag(const half_t* A, int lda, int m0, int k0) {
  const int lane = threadIdx.x & 31;
  const half_t* p = A + (size_t)(m0 + (lane & 15)) * lda + k0 + ((lane >> 4) << 3);
  union { v16h v; v8h h[2]; } u;
  u.h[0] = *(const v8h*)(p);
  u.h[1] = *(const v8h*)(p + 16);
  return u.v;
}

// B fragment (32x16 KxN) from LDS tile swizzled as [chunk16][row16][8 halves].
// lane holds column n=lane&15; K values k0 + (lane>>4)*8 + {0..7,16..23}.
static __device__ __forceinline__ v16h load_b_frag_lds(const half_t* tile, int k0) {
  const int lane = threadIdx.x & 31;
  const int row  = lane & 15;
  const int c0   = (k0 >> 3) + (lane >> 4);
  union { v16h v; v8h h[2]; } u;
  u.h[0] = *(const v8h*)(tile + ((c0    ) * 16 + row) * 8);
  u.h[1] = *(const v8h*)(tile + ((c0 + 2) * 16 + row) * 8);
  return u.v;
}

// Stage one 16xK(=128) f16 weight tile (4 KB) into LDS via the CDNA5 async
// global->LDS DMA path; each of the 256 threads copies one 16B chunk.
// LDS layout: [chunk = t&15][row = t>>4][8 halves].
static __device__ __forceinline__ void stage_b_tile(half_t* tileB, const half_t* Wz,
                                                    int ldb, int n0) {
  const int t = threadIdx.x;
  const half_t* gsrc = Wz + (size_t)(n0 + (t >> 4)) * ldb + (t & 15) * 8;
  unsigned ldst = (unsigned)(uintptr_t)(tileB + (((t & 15) * 16 + (t >> 4)) * 8));
  uint64_t ga = (uint64_t)(uintptr_t)gsrc;
  asm volatile("global_load_async_to_lds_b128 %0, %1, off" :: "v"(ldst), "v"(ga) : "memory");
  asm volatile("s_wait_asynccnt 0" ::: "memory");
  __syncthreads();
}

// ---------------------------------------------------------------------------
// GEMM: C[z] = A[z] (Mx128 f16) @ W[z]^T (Nx128 f16). Block = 256 threads =
// 8 waves; block covers 256 rows x 16 cols; weight tile staged once in LDS.
// ---------------------------------------------------------------------------
template<bool OUT16>
__global__ __launch_bounds__(256)
void k_gemm(const half_t* __restrict__ A, int lda, long sA,
            const half_t* __restrict__ W, int ldb, long sB,
            void* __restrict__ Cout, int ldc, long sC, int M)
{
  __shared__ half_t tileB[16 * 128];
  const int z = blockIdx.z;
  const half_t* Az = A + (size_t)z * sA;
  const int n0 = blockIdx.y * 16;
  stage_b_tile(tileB, W + (size_t)z * sB, ldb, n0);

  const int m0 = (blockIdx.x * 8 + (threadIdx.x >> 5)) * 32;
  v8f acc0 = {0.f,0.f,0.f,0.f,0.f,0.f,0.f,0.f};
  v8f acc1 = {0.f,0.f,0.f,0.f,0.f,0.f,0.f,0.f};

  #pragma unroll
  for (int k0 = 0; k0 < 128; k0 += 32) {
    if (k0 + 32 < 128) {  // CDNA5 global_prefetch_b8 for the next K-slab of A
      __builtin_prefetch(Az + (size_t)(m0 + (threadIdx.x & 31)) * lda + k0 + 32, 0, 1);
    }
    v16h a0 = load_a_frag(Az, lda, m0,      k0);
    v16h a1 = load_a_frag(Az, lda, m0 + 16, k0);
    v16h b  = load_b_frag_lds(tileB, k0);
    acc0 = __builtin_amdgcn_wmma_f32_16x16x32_f16(false, a0, false, b, (short)0, acc0, false, false);
    acc1 = __builtin_amdgcn_wmma_f32_16x16x32_f16(false, a1, false, b, (short)0, acc1, false, false);
  }

  const int lane = threadIdx.x & 31;
  const int n  = n0 + (lane & 15);
  const int mb = m0 + ((lane >> 4) << 3);   // C layout: VGPR r -> M = mb + r
  if constexpr (OUT16) {
    half_t* Cz = (half_t*)Cout + (size_t)z * sC;
    #pragma unroll
    for (int r = 0; r < 8; ++r) {
      Cz[(size_t)(mb + r)      * ldc + n] = (half_t)acc0[r];
      Cz[(size_t)(mb + 16 + r) * ldc + n] = (half_t)acc1[r];
    }
  } else {
    float* Cz = (float*)Cout + (size_t)z * sC;
    #pragma unroll
    for (int r = 0; r < 8; ++r) {
      Cz[(size_t)(mb + r)      * ldc + n] = acc0[r];
      Cz[(size_t)(mb + 16 + r) * ldc + n] = acc1[r];
    }
  }
}

// ---------------------------------------------------------------------------
// Final GEMM: out(B,C,H,W) = sg * (yfused @ out_w^T) + (1-sg) * x   (skip addr
// in output layout == input x addr).
// ---------------------------------------------------------------------------
__global__ __launch_bounds__(256)
void k_gemm_final(const half_t* __restrict__ A, const half_t* __restrict__ W,
                  const float* __restrict__ xin, const float* __restrict__ sgp,
                  float* __restrict__ out)
{
  __shared__ half_t tileB[16 * 128];
  const int n0 = blockIdx.y * 16;
  stage_b_tile(tileB, W, Dd, n0);

  const int m0 = (blockIdx.x * 8 + (threadIdx.x >> 5)) * 32;
  v8f acc0 = {0.f,0.f,0.f,0.f,0.f,0.f,0.f,0.f};
  v8f acc1 = {0.f,0.f,0.f,0.f,0.f,0.f,0.f,0.f};
  #pragma unroll
  for (int k0 = 0; k0 < Dd; k0 += 32) {
    v16h a0 = load_a_frag(A, Dd, m0,      k0);
    v16h a1 = load_a_frag(A, Dd, m0 + 16, k0);
    v16h b  = load_b_frag_lds(tileB, k0);
    acc0 = __builtin_amdgcn_wmma_f32_16x16x32_f16(false, a0, false, b, (short)0, acc0, false, false);
    acc1 = __builtin_amdgcn_wmma_f32_16x16x32_f16(false, a1, false, b, (short)0, acc1, false, false);
  }
  const float sg = *sgp, sk = 1.f - sg;
  const int lane = threadIdx.x & 31;
  const int c  = n0 + (lane & 15);
  const int mb = m0 + ((lane >> 4) << 3);
  #pragma unroll
  for (int r = 0; r < 8; ++r) {
    int m  = mb + r;
    size_t a0i = ((size_t)((m >> 12) * C_ + c)) * Ls + (m & 4095);
    out[a0i] = sg * acc0[r] + sk * xin[a0i];
    int m2 = m + 16;
    size_t a1i = ((size_t)((m2 >> 12) * C_ + c)) * Ls + (m2 & 4095);
    out[a1i] = sg * acc1[r] + sk * xin[a1i];
  }
}

// --------------------------- helper kernels --------------------------------
__global__ void k_f32tof16(const float* __restrict__ s, half_t* __restrict__ d, int n) {
  int i = blockIdx.x * blockDim.x + threadIdx.x;
  if (i < n) d[i] = (half_t)s[i];
}

// m_xproj_w (4,40,128) -> f16 (4,48,128) zero-padded
__global__ void k_cvt_xproj(const float* __restrict__ s, half_t* __restrict__ d) {
  int t = blockIdx.x * blockDim.x + threadIdx.x;
  if (t >= NDIR * XPN * Dd) return;
  int c = t & 127; int r = (t >> 7) % XPN; int dir = t / (XPN * Dd);
  d[t] = (r < 40) ? (half_t)s[((size_t)dir * 40 + r) * Dd + c] : (half_t)0.f;
}

// LayerNorm over C: one wave per row (b,l); x is (B,C,H,W)
__global__ __launch_bounds__(256)
void k_layernorm(const float* __restrict__ x, const float* __restrict__ g,
                 const float* __restrict__ bt, half_t* __restrict__ out)
{
  int w = blockIdx.x * (blockDim.x >> 5) + (threadIdx.x >> 5);
  if (w >= ROWS) return;
  int lane = threadIdx.x & 31;
  int b = w >> 12, l = w & 4095;
  const float* base = x + (size_t)b * C_ * Ls + l;
  float v[4], s = 0.f;
  #pragma unroll
  for (int i = 0; i < 4; ++i) { v[i] = base[(size_t)(lane + 32 * i) * Ls]; s += v[i]; }
  #pragma unroll
  for (int o = 16; o; o >>= 1) s += __shfl_xor(s, o, 32);
  float mu = s * (1.f / 128.f), q = 0.f;
  #pragma unroll
  for (int i = 0; i < 4; ++i) { float d = v[i] - mu; q += d * d; }
  #pragma unroll
  for (int o = 16; o; o >>= 1) q += __shfl_xor(q, o, 32);
  float rs = rsqrtf(q * (1.f / 128.f) + 1e-5f);
  #pragma unroll
  for (int i = 0; i < 4; ++i) {
    int c = lane + 32 * i;
    out[(size_t)w * Dd + c] = (half_t)((v[i] - mu) * rs * g[c] + bt[c]);
  }
}

// Build 4 direction sequences U[dir] from xp (fwd / rev / transpose / rev-transpose)
__global__ void k_gather_dirs(const half_t* __restrict__ xp, half_t* __restrict__ U) {
  int t = blockIdx.x * blockDim.x + threadIdx.x;          // (dir,row,chunk of 8 halves)
  if (t >= NDIR * ROWS * 16) return;
  int chunk = t & 15;
  int rr = t >> 4;
  int dir = rr >> 15;               // ROWS = 2^15
  int row = rr & 32767;
  int b = row >> 12, l = row & 4095;
  int src;
  if      (dir == 0) src = l;
  else if (dir == 1) src = 4095 - l;
  else if (dir == 2) src = ((l & 63) << 6) | (l >> 6);
  else { int lr = 4095 - l; src = ((lr & 63) << 6) | (lr >> 6); }
  *(v8h*)(U + (size_t)t * 8) =
      *(const v8h*)(xp + ((size_t)(b * Ls + src) * Dd) + chunk * 8);
}

// depthwise causal conv (K=4) + SiLU : xz[:, :D] -> xh
__global__ void k_conv_silu(const half_t* __restrict__ xz, const float* __restrict__ cw,
                            const float* __restrict__ cb, half_t* __restrict__ xh)
{
  int t = blockIdx.x * blockDim.x + threadIdx.x;          // (dir*ROWS+row)*128 + d
  if (t >= NDIR * ROWS * Dd) return;
  int d = t & 127;
  long fr = t >> 7;
  int dir = (int)(fr >> 15);
  int l = (int)(fr & 4095);
  float acc = cb[dir * Dd + d];
  const float* wk = cw + ((size_t)dir * Dd + d) * 4;
  #pragma unroll
  for (int k = 0; k < 4; ++k) {
    int lp = l - 3 + k;
    if (lp >= 0) acc += wk[k] * (float)xz[(fr - (3 - k)) * 256 + d];
  }
  xh[(size_t)fr * Dd + d] = (half_t)(acc / (1.f + expf(-acc)));
}

// dt = softplus(x_dbl[:, :R] @ dt_w^T + dt_b)
__global__ void k_dtproj(const float* __restrict__ xdbl, const float* __restrict__ dtw,
                         const float* __restrict__ dtb, float* __restrict__ dt)
{
  int t = blockIdx.x * blockDim.x + threadIdx.x;
  if (t >= NDIR * ROWS * Dd) return;
  int d = t & 127;
  long fr = t >> 7;
  int dir = (int)(fr >> 15);
  const float* xr = xdbl + fr * XPN;
  const float* wv = dtw + ((size_t)dir * Dd + d) * Rr;
  float a = dtb[dir * Dd + d];
  #pragma unroll
  for (int r = 0; r < Rr; ++r) a += xr[r] * wv[r];
  dt[t] = (a > 20.f) ? a : log1pf(expf(a));
}

// selective scan: lane = (n, d-sub), wave handles (dir, b, d-pair); shuffle-xor
// reduction over the 16 n-lanes; epilogue: (+ x*Dp) * silu(z)
__global__ __launch_bounds__(256)
void k_scan(const half_t* __restrict__ xh, const float* __restrict__ dt,
            const float* __restrict__ xdbl, const half_t* __restrict__ xz,
            const float* __restrict__ Alog, const float* __restrict__ Dp,
            half_t* __restrict__ yact)
{
  int wave = blockIdx.x * (blockDim.x >> 5) + (threadIdx.x >> 5);   // 0..2047
  int lane = threadIdx.x & 31;
  int n = lane & 15, dsub = lane >> 4;
  int dir = wave >> 9;
  int rem = wave & 511;
  int b = rem >> 6;
  int d = ((rem & 63) << 1) + dsub;
  float A  = -expf(Alog[((size_t)dir * Dd + d) * Nn + n]);
  float Dv = Dp[dir * Dd + d];
  long fr0 = (long)dir * ROWS + (long)b * Ls;
  float h = 0.f;
  for (int l = 0; l < Ls; ++l) {
    long fr = fr0 + l;
    float dtv = dt[fr * Dd + d];
    float xv  = (float)xh[fr * Dd + d];
    float Bt  = xdbl[fr * XPN + Rr + n];
    float Ct  = xdbl[fr * XPN + Rr + Nn + n];
    h = expf(dtv * A) * h + (dtv * xv) * Bt;
    float y = h * Ct;
    y += __shfl_xor(y, 1, 32);
    y += __shfl_xor(y, 2, 32);
    y += __shfl_xor(y, 4, 32);
    y += __shfl_xor(y, 8, 32);
    if (n == 0) {
      float z = (float)xz[fr * 256 + Dd + d];
      yact[fr * Dd + d] = (half_t)((y + xv * Dv) * (z / (1.f + expf(-z))));
    }
  }
}

// softmax-gated fusion with inverse direction maps -> f16 for final GEMM
__global__ void k_fuse(const float* __restrict__ Y, const float* __restrict__ fg,
                       half_t* __restrict__ yf)
{
  int t = blockIdx.x * blockDim.x + threadIdx.x;          // row*128 + c
  if (t >= ROWS * Dd) return;
  int c = t & 127; int row = t >> 7;
  int b = row >> 12, l = row & 4095;
  float g0 = fg[0], g1 = fg[1], g2 = fg[2], g3 = fg[3];
  float mx = fmaxf(fmaxf(g0, g1), fmaxf(g2, g3));
  float e0 = expf(g0 - mx), e1 = expf(g1 - mx), e2 = expf(g2 - mx), e3 = expf(g3 - mx);
  float inv = 1.f / (e0 + e1 + e2 + e3);
  int s2 = ((l & 63) << 6) | (l >> 6);
  long bl = (long)b * Ls;
  float v = e0 * Y[((long)0 * ROWS + bl + l)          * Dd + c]
          + e1 * Y[((long)1 * ROWS + bl + 4095 - l)   * Dd + c]
          + e2 * Y[((long)2 * ROWS + bl + s2)         * Dd + c]
          + e3 * Y[((long)3 * ROWS + bl + 4095 - s2)  * Dd + c];
  yf[t] = (half_t)(v * inv);
}

// ---------------------------------------------------------------------------
extern "C" void kernel_launch(void* const* d_in, const int* in_sizes, int n_in,
                              void* d_out, int out_size, void* d_ws, size_t ws_size,
                              hipStream_t stream) {
  const float* x      = (const float*)d_in[0];
  const float* ln_g   = (const float*)d_in[1];
  const float* ln_b   = (const float*)d_in[2];
  const float* in_w   = (const float*)d_in[3];
  const float* m_in_w = (const float*)d_in[4];
  const float* m_cw   = (const float*)d_in[5];
  const float* m_cb   = (const float*)d_in[6];
  const float* m_xpw  = (const float*)d_in[7];
  const float* m_dtw  = (const float*)d_in[8];
  const float* m_dtb  = (const float*)d_in[9];
  const float* m_Alog = (const float*)d_in[10];
  const float* m_D    = (const float*)d_in[11];
  const float* m_ow   = (const float*)d_in[12];
  const float* f_gate = (const float*)d_in[13];
  const float* out_w  = (const float*)d_in[14];
  const float* s_gate = (const float*)d_in[15];
  float* out = (float*)d_out;

  char* ws = (char*)d_ws;
  size_t off = 0;
  auto take = [&](size_t bytes) { void* p = ws + off; off = (off + bytes + 255) & ~(size_t)255; return p; };

  half_t* xnorm = (half_t*)take((size_t)ROWS * Dd * 2);
  half_t* xp    = (half_t*)take((size_t)ROWS * Dd * 2);
  half_t* U     = (half_t*)take((size_t)NDIR * ROWS * Dd * 2);
  half_t* xz    = (half_t*)take((size_t)NDIR * ROWS * 256 * 2);
  half_t* xh    = (half_t*)take((size_t)NDIR * ROWS * Dd * 2);
  float*  xdbl  = (float*) take((size_t)NDIR * ROWS * XPN * 4);
  float*  dtb_  = (float*) take((size_t)NDIR * ROWS * Dd * 4);
  half_t* yact  = (half_t*)take((size_t)NDIR * ROWS * Dd * 2);
  float*  ydir  = (float*) take((size_t)NDIR * ROWS * Dd * 4);
  half_t* yfus  = (half_t*)take((size_t)ROWS * Dd * 2);
  half_t* w_in  = (half_t*)take((size_t)Dd * Dd * 2);
  half_t* w_min = (half_t*)take((size_t)NDIR * 256 * Dd * 2);
  half_t* w_xp  = (half_t*)take((size_t)NDIR * XPN * Dd * 2);
  half_t* w_mo  = (half_t*)take((size_t)NDIR * Dd * Dd * 2);
  half_t* w_out = (half_t*)take((size_t)Dd * Dd * 2);

  // weight conversions
  k_f32tof16<<<(Dd*Dd + 255) / 256, 256, 0, stream>>>(in_w,  w_in,  Dd * Dd);
  k_f32tof16<<<(NDIR*256*Dd + 255) / 256, 256, 0, stream>>>(m_in_w, w_min, NDIR * 256 * Dd);
  k_f32tof16<<<(NDIR*Dd*Dd + 255) / 256, 256, 0, stream>>>(m_ow,  w_mo,  NDIR * Dd * Dd);
  k_f32tof16<<<(Dd*Dd + 255) / 256, 256, 0, stream>>>(out_w, w_out, Dd * Dd);
  k_cvt_xproj<<<(NDIR*XPN*Dd + 255) / 256, 256, 0, stream>>>(m_xpw, w_xp);

  // LayerNorm -> xnorm (f16)
  k_layernorm<<<ROWS / 8, 256, 0, stream>>>(x, ln_g, ln_b, xnorm);

  // xp = xnorm @ in_w^T  (32768x128x128)
  k_gemm<true><<<dim3(ROWS / 256, Dd / 16, 1), 256, 0, stream>>>(
      xnorm, Dd, 0, w_in, Dd, 0, xp, Dd, 0, ROWS);

  // build 4 direction sequences
  k_gather_dirs<<<(NDIR * ROWS * 16) / 256, 256, 0, stream>>>(xp, U);

  // xz[d] = U[d] @ m_in_w[d]^T  (32768x128x256, 4 dirs)
  k_gemm<true><<<dim3(ROWS / 256, 256 / 16, NDIR), 256, 0, stream>>>(
      U, Dd, (long)ROWS * Dd, w_min, Dd, (long)256 * Dd,
      xz, 256, (long)ROWS * 256, ROWS);

  // depthwise conv + SiLU -> xh
  k_conv_silu<<<(NDIR * ROWS * Dd) / 256, 256, 0, stream>>>(xz, m_cw, m_cb, xh);

  // x_dbl[d] = xh[d] @ xproj_pad[d]^T  (N=48, f32 out)
  k_gemm<false><<<dim3(ROWS / 256, XPN / 16, NDIR), 256, 0, stream>>>(
      xh, Dd, (long)ROWS * Dd, w_xp, Dd, (long)XPN * Dd,
      xdbl, XPN, (long)ROWS * XPN, ROWS);

  // dt projection + softplus
  k_dtproj<<<(NDIR * ROWS * Dd) / 256, 256, 0, stream>>>(xdbl, m_dtw, m_dtb, dtb_);

  // selective scan (+D skip, * silu(z)) -> yact (f16)
  k_scan<<<256, 256, 0, stream>>>(xh, dtb_, xdbl, xz, m_Alog, m_D, yact);

  // ydir[d] = yact[d] @ m_out_w[d]^T (f32 out)
  k_gemm<false><<<dim3(ROWS / 256, Dd / 16, NDIR), 256, 0, stream>>>(
      yact, Dd, (long)ROWS * Dd, w_mo, Dd, (long)Dd * Dd,
      ydir, Dd, (long)ROWS * Dd, ROWS);

  // gated fusion with inverse direction maps -> yfused (f16)
  k_fuse<<<(ROWS * Dd) / 256, 256, 0, stream>>>(ydir, f_gate, yfus);

  // final out = sg * (yfused @ out_w^T) + (1-sg) * skip, stored as (B,C,H,W)
  k_gemm_final<<<dim3(ROWS / 256, Dd / 16), 256, 0, stream>>>(
      yfus, w_out, x, s_gate, out);
}
